// RoadGIN_62577673503437
// MI455X (gfx1250) — matrix operand order repaired and checked by reading, not compiled, not run on hardware
//
#include <hip/hip_runtime.h>
#include <hip/hip_bf16.h>

#define N_NODES 50000
#define N_EDGES 600000
#define DIM     128
#define DIM2    256
#define DEPTH   3
#define BN_EPS  1e-5f

typedef __attribute__((ext_vector_type(16))) __bf16 v16bf;
typedef __attribute__((ext_vector_type(8)))  float  v8f;

// ---------------------------------------------------------------- utilities

__global__ void k_zero_out(float* __restrict__ out, int n) {
    int i = blockIdx.x * blockDim.x + threadIdx.x;
    if (i < n) out[i] = 0.0f;
}

// Repack one layer's W1 [128x256] and W2 [256x128] (row-major f32) into
// bf16 WMMA B-fragment order: element (((ct*KC + kc)*32 + lane)*16 + j) =
// W[kc*32 + (lane>>4)*8 + j + (j&8)][ct*16 + (lane&15)], so each (ct,kc,lane)
// fragment slice is 32 contiguous bytes (two b128 loads per lane).
// Also zeroes the 768-float BN statistics block.
__global__ void k_pack(const float* __restrict__ W1, const float* __restrict__ W2,
                       __bf16* __restrict__ W1p, __bf16* __restrict__ W2p,
                       float* __restrict__ stats) {
    int t = blockIdx.x * blockDim.x + threadIdx.x;
    if (t < 768) stats[t] = 0.0f;
    if (t < 32768) {            // W1: 16 col-tiles x 4 k-chunks
        int j = t & 15, lane = (t >> 4) & 31, kc = (t >> 9) & 3, ct = t >> 11;
        int k = kc * 32 + (lane >> 4) * 8 + j + (j & 8);
        W1p[t] = (__bf16)W1[k * DIM2 + ct * 16 + (lane & 15)];
    } else if (t < 65536) {     // W2: 8 col-tiles x 8 k-chunks
        int t2 = t - 32768;
        int j = t2 & 15, lane = (t2 >> 4) & 31, kc = (t2 >> 9) & 7, ct = t2 >> 12;
        int k = kc * 32 + (lane >> 4) * 8 + j + (j & 8);
        W2p[t2] = (__bf16)W2[k * DIM + ct * 16 + (lane & 15)];
    }
}

__global__ void k_init_h(const float* __restrict__ x, const float* __restrict__ eps,
                         int layer, float* __restrict__ h) {
    int i = blockIdx.x * blockDim.x + threadIdx.x;
    if (i < N_NODES * DIM) h[i] = (1.0f + eps[layer]) * x[i];
}

// One wave (32 lanes) per edge; lane covers 4 consecutive features.
__global__ void k_edge_agg(const float* __restrict__ x, const long long* __restrict__ ei,
                           float* __restrict__ h) {
    long long tid = (long long)blockIdx.x * blockDim.x + threadIdx.x;
    int e    = (int)(tid >> 5);
    int lane = (int)(tid & 31);
    if (e < N_EDGES) {
        int s = (int)ei[e];
        int d = (int)ei[N_EDGES + e];
        const float4 v = *(const float4*)(x + (size_t)s * DIM + lane * 4);
        float* p = h + (size_t)d * DIM + lane * 4;
        atomicAdd(p + 0, v.x);
        atomicAdd(p + 1, v.y);
        atomicAdd(p + 2, v.z);
        atomicAdd(p + 3, v.w);
    }
}

// ---------------------------------------------------------------- GEMM1
// h1[N,256] = h[N,128] @ W1 + b1 ; accumulate per-column sum / sumsq.
// Block = 128 threads = 4 waves; each wave owns a 16-row tile x all 256 cols.
__global__ __launch_bounds__(128) void k_gemm1(
        const float* __restrict__ h, const __bf16* __restrict__ W1p,
        const float* __restrict__ b1, float* __restrict__ h1,
        float* __restrict__ gsum, float* __restrict__ gsq) {
    __shared__ float s_sum[DIM2], s_sq[DIM2];
    for (int i = threadIdx.x; i < DIM2; i += 128) { s_sum[i] = 0.f; s_sq[i] = 0.f; }
    __syncthreads();

    const int wave = threadIdx.x >> 5, lane = threadIdx.x & 31;
    const int m0 = blockIdx.x * 64 + wave * 16;       // wave-uniform guard
    if (m0 < N_NODES) {
        const int mrow = m0 + (lane & 15);
        const int kh   = lane >> 4;                   // K-group half select
        const float* rp = h + (size_t)mrow * DIM;
        v16bf a[4];
#pragma unroll
        for (int kc = 0; kc < 4; ++kc) {              // K = 128 -> 4 chunks of 32
            const int kb = kc * 32 + kh * 8;
            float tmp[16];
            *(float4*)(tmp + 0)  = *(const float4*)(rp + kb);
            *(float4*)(tmp + 4)  = *(const float4*)(rp + kb + 4);
            *(float4*)(tmp + 8)  = *(const float4*)(rp + kb + 16);
            *(float4*)(tmp + 12) = *(const float4*)(rp + kb + 20);
#pragma unroll
            for (int j = 0; j < 16; ++j) a[kc][j] = (__bf16)tmp[j];
        }
        for (int ct = 0; ct < 16; ++ct) {             // 256 cols -> 16 tiles
            v8f acc = {};
#pragma unroll
            for (int kc = 0; kc < 4; ++kc) {
                v16bf b = *(const v16bf*)(W1p + (((ct * 4 + kc) * 32 + lane) << 4));
                acc = __builtin_amdgcn_wmma_f32_16x16x32_bf16(
                        false, a[kc], false, b, (short)0, acc, false, false);
            }
            const int n = ct * 16 + (lane & 15);
            const float bias = b1[n];
            float s = 0.f, ss = 0.f;
            float* op = h1 + (size_t)(m0 + kh * 8) * DIM2 + n;
#pragma unroll
            for (int r = 0; r < 8; ++r) {
                float v = acc[r] + bias;
                op[(size_t)r * DIM2] = v;
                s += v; ss += v * v;
            }
            atomicAdd(&s_sum[n], s);
            atomicAdd(&s_sq[n], ss);
        }
    }
    __syncthreads();
    for (int i = threadIdx.x; i < DIM2; i += 128) {
        atomicAdd(&gsum[i], s_sum[i]);
        atomicAdd(&gsq[i],  s_sq[i]);
    }
}

// Fold BN stats + gamma/beta into per-column scale/shift.
__global__ void k_finalize(const float* __restrict__ gsum, const float* __restrict__ gsq,
                           const float* __restrict__ g, const float* __restrict__ beta,
                           float* __restrict__ scale, float* __restrict__ shift, int C) {
    int c = threadIdx.x;
    if (c < C) {
        const float inv = 1.0f / (float)N_NODES;
        float mu  = gsum[c] * inv;
        float var = gsq[c] * inv - mu * mu;
        float sc  = g[c] * rsqrtf(var + BN_EPS);
        scale[c] = sc;
        shift[c] = beta[c] - mu * sc;
    }
}

// ---------------------------------------------------------------- GEMM2
// h2[N,128] = relu(BN1(h1)) @ W2 + b2, BN1 fused into the A-fragment load.
__global__ __launch_bounds__(128) void k_gemm2(
        const float* __restrict__ h1, const __bf16* __restrict__ W2p,
        const float* __restrict__ b2, const float* __restrict__ scale1,
        const float* __restrict__ shift1, float* __restrict__ h2,
        float* __restrict__ gsum, float* __restrict__ gsq) {
    __shared__ float s_sc[DIM2], s_sh[DIM2];
    __shared__ float s_sum[DIM], s_sq[DIM];
    for (int i = threadIdx.x; i < DIM2; i += 128) { s_sc[i] = scale1[i]; s_sh[i] = shift1[i]; }
    if (threadIdx.x < DIM) { s_sum[threadIdx.x] = 0.f; s_sq[threadIdx.x] = 0.f; }
    __syncthreads();

    const int wave = threadIdx.x >> 5, lane = threadIdx.x & 31;
    const int m0 = blockIdx.x * 64 + wave * 16;       // wave-uniform guard
    if (m0 < N_NODES) {
        const int mrow = m0 + (lane & 15);
        const int kh   = lane >> 4;
        const float* rp = h1 + (size_t)mrow * DIM2;
        v16bf a[8];
#pragma unroll
        for (int kc = 0; kc < 8; ++kc) {              // K = 256 -> 8 chunks
            const int kb = kc * 32 + kh * 8;
            float tmp[16];
            *(float4*)(tmp + 0)  = *(const float4*)(rp + kb);
            *(float4*)(tmp + 4)  = *(const float4*)(rp + kb + 4);
            *(float4*)(tmp + 8)  = *(const float4*)(rp + kb + 16);
            *(float4*)(tmp + 12) = *(const float4*)(rp + kb + 20);
#pragma unroll
            for (int j = 0; j < 8; ++j) {
                int k = kb + j;
                a[kc][j]     = (__bf16)fmaxf(fmaf(tmp[j],     s_sc[k],      s_sh[k]),      0.f);
                a[kc][8 + j] = (__bf16)fmaxf(fmaf(tmp[8 + j], s_sc[k + 16], s_sh[k + 16]), 0.f);
            }
        }
        for (int ct = 0; ct < 8; ++ct) {              // 128 cols -> 8 tiles
            v8f acc = {};
#pragma unroll
            for (int kc = 0; kc < 8; ++kc) {
                v16bf b = *(const v16bf*)(W2p + (((ct * 8 + kc) * 32 + lane) << 4));
                acc = __builtin_amdgcn_wmma_f32_16x16x32_bf16(
                        false, a[kc], false, b, (short)0, acc, false, false);
            }
            const int n = ct * 16 + (lane & 15);
            const float bias = b2[n];
            float s = 0.f, ss = 0.f;
            float* op = h2 + (size_t)(m0 + kh * 8) * DIM + n;
#pragma unroll
            for (int r = 0; r < 8; ++r) {
                float v = acc[r] + bias;
                op[(size_t)r * DIM] = v;
                s += v; ss += v * v;
            }
            atomicAdd(&s_sum[n], s);
            atomicAdd(&s_sq[n], ss);
        }
    }
    __syncthreads();
    if (threadIdx.x < DIM) {
        atomicAdd(&gsum[threadIdx.x], s_sum[threadIdx.x]);
        atomicAdd(&gsq[threadIdx.x],  s_sq[threadIdx.x]);
    }
}

// BN2 + ReLU -> next-layer x, and running depth-max into out.
__global__ void k_apply2(const float* __restrict__ h2, const float* __restrict__ scale,
                         const float* __restrict__ shift, float* __restrict__ xnext,
                         float* __restrict__ out) {
    int i = blockIdx.x * blockDim.x + threadIdx.x;
    if (i < N_NODES * DIM) {
        int c = i & (DIM - 1);
        float v = fmaxf(fmaf(h2[i], scale[c], shift[c]), 0.f);
        xnext[i] = v;
        out[i] = fmaxf(out[i], v);
    }
}

// ---------------------------------------------------------------- launch

extern "C" void kernel_launch(void* const* d_in, const int* in_sizes, int n_in,
                              void* d_out, int out_size, void* d_ws, size_t ws_size,
                              hipStream_t stream) {
    const float*     x0  = (const float*)d_in[0];
    const long long* ei  = (const long long*)d_in[1];
    const float*     eps = (const float*)d_in[2];
    const float*     W1  = (const float*)d_in[3];
    const float*     b1  = (const float*)d_in[4];
    const float*     g1  = (const float*)d_in[5];
    const float*     be1 = (const float*)d_in[6];
    const float*     W2  = (const float*)d_in[7];
    const float*     b2  = (const float*)d_in[8];
    const float*     g2  = (const float*)d_in[9];
    const float*     be2 = (const float*)d_in[10];
    float* out = (float*)d_out;

    char* ws = (char*)d_ws;
    size_t off = 0;
    auto take = [&](size_t bytes) -> void* {
        void* p = ws + off;
        off += (bytes + 255) & ~(size_t)255;
        return p;
    };
    float*  h      = (float*)take(sizeof(float) * (size_t)N_NODES * DIM);
    float*  h1     = (float*)take(sizeof(float) * (size_t)N_NODES * DIM2);
    float*  h2     = (float*)take(sizeof(float) * (size_t)N_NODES * DIM);
    float*  xcur   = (float*)take(sizeof(float) * (size_t)N_NODES * DIM);
    __bf16* W1p    = (__bf16*)take(sizeof(__bf16) * 32768);
    __bf16* W2p    = (__bf16*)take(sizeof(__bf16) * 32768);
    float*  stats  = (float*)take(sizeof(float) * 768);   // sum1[256] sq1[256] sum2[128] sq2[128]
    float*  scale1 = (float*)take(sizeof(float) * DIM2);
    float*  shift1 = (float*)take(sizeof(float) * DIM2);
    float*  scale2 = (float*)take(sizeof(float) * DIM);
    float*  shift2 = (float*)take(sizeof(float) * DIM);

    const int ND       = N_NODES * DIM;
    const int elemBlks = (ND + 255) / 256;
    const int edgeBlks = (N_EDGES * 32 + 255) / 256;
    const int gemmBlks = (N_NODES + 63) / 64;

    k_zero_out<<<elemBlks, 256, 0, stream>>>(out, ND);

    for (int layer = 0; layer < DEPTH; ++layer) {
        const float* xin = (layer == 0) ? x0 : xcur;
        k_pack<<<256, 256, 0, stream>>>(W1 + (size_t)layer * DIM * DIM2,
                                        W2 + (size_t)layer * DIM2 * DIM,
                                        W1p, W2p, stats);
        k_init_h<<<elemBlks, 256, 0, stream>>>(xin, eps, layer, h);
        k_edge_agg<<<edgeBlks, 256, 0, stream>>>(xin, ei, h);
        k_gemm1<<<gemmBlks, 128, 0, stream>>>(h, W1p, b1 + (size_t)layer * DIM2,
                                              h1, stats, stats + 256);
        k_finalize<<<1, 256, 0, stream>>>(stats, stats + 256,
                                          g1 + (size_t)layer * DIM2, be1 + (size_t)layer * DIM2,
                                          scale1, shift1, DIM2);
        k_gemm2<<<gemmBlks, 128, 0, stream>>>(h1, W2p, b2 + (size_t)layer * DIM,
                                              scale1, shift1, h2, stats + 512, stats + 640);
        k_finalize<<<1, 128, 0, stream>>>(stats + 512, stats + 640,
                                          g2 + (size_t)layer * DIM, be2 + (size_t)layer * DIM,
                                          scale2, shift2, DIM);
        k_apply2<<<elemBlks, 256, 0, stream>>>(h2, scale2, shift2, xcur, out);
    }
}
